// FF_Block_11708080849221
// MI455X (gfx1250) — matrix operand order repaired and checked by reading, not compiled, not run on hardware
//
#include <hip/hip_runtime.h>
#include <hip/hip_bf16.h>

// ---------------------------------------------------------------------------
// SNN Leaky layer: cur = x @ W^T  (M=T*B=8192, N=1024, K=1024), then a
// sequential leaky-integrate-and-fire scan over T.
//  K1: f32 -> f16 conversion of x and W into workspace
//  K2: WMMA f16 GEMM (v_wmma_f32_16x16x32_f16), f32 accumulate -> cur in ws
//  K3: leaky scan, coalesced over B*N, writes spk_rec then mem_rec to d_out
// ---------------------------------------------------------------------------

typedef __attribute__((ext_vector_type(16))) _Float16 v16h;
typedef __attribute__((ext_vector_type(8)))  _Float16 v8h;
typedef __attribute__((ext_vector_type(8)))  float    v8f;
typedef __attribute__((ext_vector_type(4)))  float    v4f;
typedef __attribute__((ext_vector_type(4)))  _Float16 v4h;

union AFrag { v16h v; v8h h8[2]; };

#define GEMM_M 8192   // T*B
#define GEMM_N 1024   // N_out
#define GEMM_K 1024   // N_in
#define SCAN_T 128
#define SCAN_BN 65536 // B*N_out

// ---------------- f32 -> f16 conversion (vectorized x4) --------------------
__global__ void snn_cvt_f16(const float* __restrict__ src,
                            _Float16* __restrict__ dst, int n4) {
  int i = blockIdx.x * blockDim.x + threadIdx.x;
  if (i >= n4) return;
  v4f s = *(const v4f*)(src + (size_t)i * 4);
  v4h d;
  d.x = (_Float16)s.x; d.y = (_Float16)s.y;
  d.z = (_Float16)s.z; d.w = (_Float16)s.w;
  *(v4h*)(dst + (size_t)i * 4) = d;
}

// ---------------- WMMA GEMM: cur[m][n] = sum_k xh[m][k] * wh[n][k] ---------
// Block: 256 threads = 8 waves, arranged 4 (M) x 2 (N); each wave owns a
// 32x32 tile (2x2 grid of 16x16 WMMA tiles) -> block tile = 128 (M) x 64 (N).
// grid = (N/64, M/128) = (16, 64).
__global__ void __launch_bounds__(256)
snn_wmma_gemm(const _Float16* __restrict__ xh,   // M x K row-major
              const _Float16* __restrict__ wh,   // N x K row-major (W)
              float* __restrict__ cur) {         // M x N row-major
  const int lane   = threadIdx.x & 31;
  const int wave   = threadIdx.x >> 5;
  const int wave_m = wave >> 1;        // 0..3
  const int wave_n = wave & 1;         // 0..1

  const int m0 = blockIdx.y * 128 + wave_m * 32;  // wave M base
  const int n0 = blockIdx.x * 64  + wave_n * 32;  // wave N base

  const int lhalf = lane & 15;
  const int hi    = lane >> 4;         // 0: lanes 0-15, 1: lanes 16-31

  // A-fragment rows for the two M sub-tiles (lane -> row m0 + lhalf [+16])
  const _Float16* arow0 = xh + (size_t)(m0 + lhalf)      * GEMM_K;
  const _Float16* arow1 = xh + (size_t)(m0 + 16 + lhalf) * GEMM_K;
  // B-fragment rows of W for the two N sub-tiles (lane -> column n0 + lhalf)
  const _Float16* brow0 = wh + (size_t)(n0 + lhalf)      * GEMM_K;
  const _Float16* brow1 = wh + (size_t)(n0 + 16 + lhalf) * GEMM_K;

  v8f acc00 = {}, acc01 = {}, acc10 = {}, acc11 = {};

  // A 16x32 f16, k order used here (contiguous, not AMD's table): lanes 0-15 hold K kk+0..15 of row lhalf,
  //                     lanes 16-31 hold K kk+16..31
  // B 32x16 f16, same k order as A: lanes 0-15 hold K kk+0..15 of column lhalf,
  //                     lanes 16-31 hold K kk+16..31
  const int a_off = hi * 16;   // 0 or 16
  const int b_off = hi * 16;   // 0 or 16

  #pragma unroll 2
  for (int kk = 0; kk < GEMM_K; kk += 32) {
    AFrag a0, a1;
    a0.h8[0] = *(const v8h*)(arow0 + kk + a_off);
    a0.h8[1] = *(const v8h*)(arow0 + kk + 8 + a_off);
    a1.h8[0] = *(const v8h*)(arow1 + kk + a_off);
    a1.h8[1] = *(const v8h*)(arow1 + kk + 8 + a_off);

    v16h b0 = *(const v16h*)(brow0 + kk + b_off);
    v16h b1 = *(const v16h*)(brow1 + kk + b_off);

    acc00 = __builtin_amdgcn_wmma_f32_16x16x32_f16(false, a0.v, false, b0,
                                                   (short)0, acc00, false, false);
    acc01 = __builtin_amdgcn_wmma_f32_16x16x32_f16(false, a0.v, false, b1,
                                                   (short)0, acc01, false, false);
    acc10 = __builtin_amdgcn_wmma_f32_16x16x32_f16(false, a1.v, false, b0,
                                                   (short)0, acc10, false, false);
    acc11 = __builtin_amdgcn_wmma_f32_16x16x32_f16(false, a1.v, false, b1,
                                                   (short)0, acc11, false, false);
  }

  // C/D 16x16 f32 layout: VGPR r -> row r (lanes 0-15) / row r+8 (lanes 16-31),
  // column = lane%16.
  const int crow = hi * 8;
  #pragma unroll
  for (int r = 0; r < 8; ++r) {
    cur[(size_t)(m0 + crow + r)      * GEMM_N + (n0 + lhalf)]      = acc00[r];
    cur[(size_t)(m0 + crow + r)      * GEMM_N + (n0 + 16 + lhalf)] = acc01[r];
    cur[(size_t)(m0 + 16 + crow + r) * GEMM_N + (n0 + lhalf)]      = acc10[r];
    cur[(size_t)(m0 + 16 + crow + r) * GEMM_N + (n0 + 16 + lhalf)] = acc11[r];
  }
}

// ---------------- Leaky-integrate-and-fire scan over T ---------------------
// snntorch Leaky, reset_mechanism='subtract', beta=0.9, threshold=1.0:
//   reset = H(mem - 1);  mem = 0.9*mem + cur_t - reset;  spk = H(mem - 1)
__global__ void snn_leaky_scan(const float* __restrict__ cur,
                               float* __restrict__ spk_rec,
                               float* __restrict__ mem_rec) {
  int idx = blockIdx.x * blockDim.x + threadIdx.x;
  if (idx >= SCAN_BN) return;
  float mem = 0.0f;
  #pragma unroll 4
  for (int t = 0; t < SCAN_T; ++t) {
    float c = cur[(size_t)t * SCAN_BN + idx];
    float reset = (mem > 1.0f) ? 1.0f : 0.0f;       // H(mem_prev - thr)
    mem = 0.9f * mem + c - reset;
    float s = (mem > 1.0f) ? 1.0f : 0.0f;           // H(mem_new - thr)
    spk_rec[(size_t)t * SCAN_BN + idx] = s;
    mem_rec[(size_t)t * SCAN_BN + idx] = mem;
  }
}

// ---------------------------------------------------------------------------
extern "C" void kernel_launch(void* const* d_in, const int* in_sizes, int n_in,
                              void* d_out, int out_size, void* d_ws, size_t ws_size,
                              hipStream_t stream) {
  const float* x = (const float*)d_in[0];   // (T,B,N_in)  = 8,388,608 f32
  const float* W = (const float*)d_in[1];   // (N_out,N_in)= 1,048,576 f32

  // Workspace layout (all 32B aligned):
  //   [0, 32MB)          cur   : f32, M x N
  //   [32MB, 48MB)       x_f16 : M x K halfs
  //   [48MB, 50MB)       W_f16 : N x K halfs
  const size_t curBytes = (size_t)GEMM_M * GEMM_N * sizeof(float);      // 33,554,432
  const size_t xhBytes  = (size_t)GEMM_M * GEMM_K * sizeof(_Float16);   // 16,777,216
  float*     cur = (float*)d_ws;
  _Float16*  xh  = (_Float16*)((char*)d_ws + curBytes);
  _Float16*  wh  = (_Float16*)((char*)d_ws + curBytes + xhBytes);

  float* spk_rec = (float*)d_out;                                   // (T,B,N)
  float* mem_rec = (float*)d_out + (size_t)SCAN_T * SCAN_BN;        // (T,B,N)

  // 1) convert inputs to f16
  {
    int n4 = (GEMM_M * GEMM_K) / 4;
    snn_cvt_f16<<<(n4 + 255) / 256, 256, 0, stream>>>(x, xh, n4);
  }
  {
    int n4 = (GEMM_N * GEMM_K) / 4;
    snn_cvt_f16<<<(n4 + 255) / 256, 256, 0, stream>>>(W, wh, n4);
  }

  // 2) WMMA GEMM: grid (N/64, M/128) = (16, 64), block 256 (8 waves)
  {
    dim3 grid(GEMM_N / 64, GEMM_M / 128);
    snn_wmma_gemm<<<grid, 256, 0, stream>>>(xh, wh, cur);
  }

  // 3) leaky scan: one thread per (b, n) neuron, coalesced over B*N
  snn_leaky_scan<<<SCAN_BN / 256, 256, 0, stream>>>(cur, spk_rec, mem_rec);
}
